// EncoderBlock_12352325943429
// MI455X (gfx1250) — matrix operand order repaired and checked by reading, not compiled, not run on hardware
//
#include <hip/hip_runtime.h>
#include <stdint.h>

// ---------------------------------------------------------------------------
// Types for CDNA5 WMMA (wave32). 16x16x32 bf16 -> f32.
// ---------------------------------------------------------------------------
typedef __attribute__((ext_vector_type(16))) __bf16 v16bf;
typedef __attribute__((ext_vector_type(8)))  float  v8f;

union FragBF {
    v16bf v;
    uint4 q[2];
};

__device__ __forceinline__ v8f wmma_bf16(const FragBF& a, const FragBF& b, v8f c) {
    return __builtin_amdgcn_wmma_f32_16x16x32_bf16(
        /*neg_a=*/false, a.v, /*neg_b=*/false, b.v,
        /*c_mod=*/(short)0, c, /*reuse_a=*/false, /*reuse_b=*/false);
}

__device__ __forceinline__ uint16_t f2bf(float f) {
    uint32_t u = __float_as_uint(f);
    u += 0x7FFFu + ((u >> 16) & 1u);   // round-to-nearest-even
    return (uint16_t)(u >> 16);
}

// LDS-relative byte offset of a __shared__ object (async-to-LDS wants the
// LDS offset, not the flat aperture address).
#define LDS_AS __attribute__((address_space(3)))
__device__ __forceinline__ uint32_t lds_off(void* p) {
    return (uint32_t)(uintptr_t)(LDS_AS char*)p;
}

// Per-lane async copy of 16 bytes global -> LDS (tracked by ASYNCcnt).
__device__ __forceinline__ void async_copy_b128(uint32_t lds_byte_off, const void* gptr) {
    uint64_t ga = (uint64_t)(uintptr_t)gptr;
    asm volatile("global_load_async_to_lds_b128 %0, %1, off"
                 :: "v"(lds_byte_off), "v"(ga) : "memory");
}
__device__ __forceinline__ void wait_async0() {
    asm volatile("s_wait_asynccnt 0" ::: "memory");
}

// Problem dims (compile-time constants).
#define BB 32
#define CC 512
#define LL 512
#define HH 8
#define DKK 64
#define NCONV 4
#define KW 7
#define BCL (BB * CC * LL)        // 8388608
#define CL  (CC * LL)             // 262144

// ---------------------------------------------------------------------------
// Elementwise: out = x + positional_encoding(c, l)
// ---------------------------------------------------------------------------
__global__ void posadd_kernel(const float* __restrict__ x, float* __restrict__ out) {
    int idx = blockIdx.x * 256 + threadIdx.x;
    if (idx >= BCL) return;
    int l = idx & 511;
    int c = (idx >> 9) & 511;
    float fc = (float)c;
    float freq, ph;
    if ((c & 1) == 0) { freq = __powf(10000.0f, -fc * (1.0f / 512.0f)); ph = 0.0f; }
    else              { freq = -__powf(10000.0f, (1.0f - fc) * (1.0f / 512.0f)); ph = 1.57079632679f; }
    out[idx] = x[idx] + __sinf((float)l * freq + ph);
}

// ---------------------------------------------------------------------------
// LayerNorm over [C,L] per batch. One block per batch.
// ---------------------------------------------------------------------------
__global__ __launch_bounds__(256) void ln2d_kernel(const float* __restrict__ in,
                                                   const float* __restrict__ w,
                                                   const float* __restrict__ bb,
                                                   float* __restrict__ out) {
    const int b = blockIdx.x;
    const int tid = threadIdx.x;
    const float* xb = in + (size_t)b * CL;
    __shared__ float sh[256], sh2[256];
    float s = 0.f, ss = 0.f;
    const float4* x4 = (const float4*)xb;
    for (int i = tid; i < CL / 4; i += 256) {
        float4 v = x4[i];
        s  += v.x + v.y + v.z + v.w;
        ss += v.x * v.x + v.y * v.y + v.z * v.z + v.w * v.w;
    }
    sh[tid] = s; sh2[tid] = ss;
    __syncthreads();
    for (int off = 128; off > 0; off >>= 1) {
        if (tid < off) { sh[tid] += sh[tid + off]; sh2[tid] += sh2[tid + off]; }
        __syncthreads();
    }
    float mean = sh[0] * (1.0f / (float)CL);
    float var  = sh2[0] * (1.0f / (float)CL) - mean * mean;
    float rstd = rsqrtf(var + 1e-5f);
    float4* o4 = (float4*)(out + (size_t)b * CL);
    const float4* w4 = (const float4*)w;
    const float4* b4 = (const float4*)bb;
    for (int i = tid; i < CL / 4; i += 256) {
        float4 v = x4[i], wv = w4[i], bv = b4[i], o;
        o.x = (v.x - mean) * rstd * wv.x + bv.x;
        o.y = (v.y - mean) * rstd * wv.y + bv.y;
        o.z = (v.z - mean) * rstd * wv.z + bv.z;
        o.w = (v.w - mean) * rstd * wv.w + bv.w;
        o4[i] = o;
    }
}

// ---------------------------------------------------------------------------
// Depthwise conv1d (K=7, pad 3) + bias, f32 [b][c][l] -> bf16 [b][l][c]
// ---------------------------------------------------------------------------
__global__ void dwconv_kernel(const float* __restrict__ in,
                              const float* __restrict__ w,   // [C][K]
                              const float* __restrict__ db,  // [C]
                              uint16_t* __restrict__ ybf) {
    int idx = blockIdx.x * 256 + threadIdx.x;      // idx = (b*512 + l)*512 + c
    if (idx >= BCL) return;
    int c = idx & 511;
    int l = (idx >> 9) & 511;
    int b = idx >> 18;
    const float* xr = in + ((size_t)b * CC + c) * LL;
    float acc = db[c];
#pragma unroll
    for (int k = 0; k < KW; ++k) {
        int ll = l + k - 3;
        if (ll >= 0 && ll < LL) acc += w[c * KW + k] * xr[ll];
    }
    ybf[idx] = f2bf(acc);
}

// f32 [b][c][l] -> bf16 [b][l][c]
__global__ void tconv_kernel(const float* __restrict__ in, uint16_t* __restrict__ out) {
    int idx = blockIdx.x * 256 + threadIdx.x;      // idx = (b*512 + l)*512 + c
    if (idx >= BCL) return;
    int c = idx & 511;
    int l = (idx >> 9) & 511;
    int b = idx >> 18;
    out[idx] = f2bf(in[((size_t)b * CC + c) * LL + l]);
}

__global__ void f32_to_bf16_kernel(const float* __restrict__ in, uint16_t* __restrict__ out, int n) {
    int i = blockIdx.x * 256 + threadIdx.x;
    if (i < n) out[i] = f2bf(in[i]);
}

// ---------------------------------------------------------------------------
// GEMM: D[b][m][n] = sum_k W[m][k] * X[b][n][k]   (M = N = K = 512 per batch)
// W bf16 row-major [512][512]; X bf16 [b][n][k] (k contiguous).
// Block: 256 threads = 8 waves; block tile 64x64; wave tile 16x32.
// The 64-row X strip (64 KB) is staged global->LDS with async b128 copies
// (ASYNCcnt), then B fragments are read from LDS (ds_load_b128); A/W
// fragments stream from global (L2-resident).
// EPI: 0 = f32 relu(acc+bias)+res -> outf[b][m][n]
//      1 = f32 (acc+bias)+res     -> outf[b][m][n]
//      2 = bf16 (acc+bias)        -> outb[b][n][m]   (transposed, for Q/K)
//      3 = bf16 (acc+bias)        -> outb[b][m][n]   (for V^T)
// ---------------------------------------------------------------------------
template <int EPI>
__global__ __launch_bounds__(256) void gemm512_kernel(const uint16_t* __restrict__ Wbf,
                                                      const uint16_t* __restrict__ Xbf,
                                                      const float* __restrict__ bias,
                                                      const float* __restrict__ res,
                                                      float* __restrict__ outf,
                                                      uint16_t* __restrict__ outb) {
    __shared__ uint16_t sX[64 * 512];                // 64 KB: X[n0blk..+63][0..511]

    const int lane  = threadIdx.x & 31;
    const int wave  = threadIdx.x >> 5;
    const int wm    = wave >> 1;
    const int wn    = wave & 1;
    const int nblk  = blockIdx.x * 64;
    const int m0    = blockIdx.y * 64 + wm * 16;
    const int n0    = nblk + wn * 32;
    const int b     = blockIdx.z;

    // ---- async stage of the X strip into LDS ----
    {
        const char* strip = (const char*)(Xbf + (size_t)b * 512 * 512 + (size_t)nblk * 512);
        uint32_t sbase = lds_off(sX);
        for (int i = threadIdx.x; i < 4096; i += 256)    // 4096 x 16B = 64 KB
            async_copy_b128(sbase + (uint32_t)i * 16, strip + (size_t)i * 16);
        wait_async0();
        __syncthreads();
    }

    const uint32_t* Wp = (const uint32_t*)Wbf;           // row stride 256 u32
    const uint32_t* Sx = (const uint32_t*)sX;            // row stride 256 u32

    const int mlane = lane & 15;
    const int hi    = lane >> 4;
    const int kbA   = hi ? 16 : 0;
    const int kbB   = hi ? 16 : 0;
    const int mA    = m0 + mlane;
    const int nL0   = wn * 32 + mlane;                   // LDS-local rows
    const int nL1   = nL0 + 16;

    v8f c0 = {}; v8f c1 = {};
    for (int kk = 0; kk < 512; kk += 32) {
        FragBF a, b0, b1;
        const uint32_t* ap  = Wp + (size_t)mA * 256 + ((kk + kbA) >> 1);
        a.q[0]  = *(const uint4*)ap;
        a.q[1]  = *(const uint4*)(ap + 4);
        const uint32_t* bp0 = Sx + (size_t)nL0 * 256 + ((kk + kbB) >> 1);
        b0.q[0] = *(const uint4*)bp0;
        b0.q[1] = *(const uint4*)(bp0 + 4);
        const uint32_t* bp1 = Sx + (size_t)nL1 * 256 + ((kk + kbB) >> 1);
        b1.q[0] = *(const uint4*)bp1;
        b1.q[1] = *(const uint4*)(bp1 + 4);
        c0 = wmma_bf16(a, b0, c0);
        c1 = wmma_bf16(a, b1, c1);
    }

    const int mrow = m0 + hi * 8;
#pragma unroll
    for (int t = 0; t < 2; ++t) {
        v8f acc = t ? c1 : c0;
        int nn = n0 + t * 16 + mlane;
        if (EPI == 0 || EPI == 1) {
            size_t base = (size_t)b * CL + (size_t)mrow * 512 + nn;
#pragma unroll
            for (int r = 0; r < 8; ++r) {
                float v = acc[r] + bias[mrow + r];
                if (EPI == 0) v = fmaxf(v, 0.0f);
                v += res[base + (size_t)r * 512];
                outf[base + (size_t)r * 512] = v;
            }
        } else if (EPI == 2) {
            alignas(16) uint16_t h[8];
#pragma unroll
            for (int r = 0; r < 8; ++r) h[r] = f2bf(acc[r] + bias[mrow + r]);
            *(uint4*)(outb + (size_t)b * CL + (size_t)nn * 512 + mrow) = *(const uint4*)h;
        } else {
            size_t base = (size_t)b * CL + (size_t)mrow * 512 + nn;
#pragma unroll
            for (int r = 0; r < 8; ++r) outb[base + (size_t)r * 512] = f2bf(acc[r] + bias[mrow + r]);
        }
    }
}

// ---------------------------------------------------------------------------
// Attention: per block = (32 q-rows, head h, batch b), 2 waves.
// qbf,kbf: bf16 [b][l][c]; vbfT: bf16 [b][c][l]; out attnbf: bf16 [b][l][c].
// Scores via WMMA into LDS, rowwise masked softmax (in-place f32->bf16),
// then P @ V via WMMA with A fragments read from LDS.
// ---------------------------------------------------------------------------
__global__ __launch_bounds__(64) void attn_kernel(const uint16_t* __restrict__ qbf,
                                                  const uint16_t* __restrict__ kbf,
                                                  const uint16_t* __restrict__ vbfT,
                                                  const uint8_t* __restrict__ mask,
                                                  uint16_t* __restrict__ attnbf) {
    __shared__ float sS[32 * 512];            // 64 KB

    const int lane  = threadIdx.x & 31;
    const int wave  = threadIdx.x >> 5;       // 0..1
    const int lt    = blockIdx.x;             // 0..15
    const int h     = blockIdx.y;             // 0..7
    const int b     = blockIdx.z;             // 0..31
    const int lq0   = lt * 32 + wave * 16;
    const int mlane = lane & 15;
    const int hi    = lane >> 4;
    const int kbA   = hi ? 16 : 0;
    const int kbB   = hi ? 16 : 0;
    const int hc    = h * DKK;

    const uint32_t* Qp = (const uint32_t*)qbf  + (size_t)b * 512 * 256;
    const uint32_t* Kp = (const uint32_t*)kbf  + (size_t)b * 512 * 256;
    const uint32_t* Vp = (const uint32_t*)vbfT + (size_t)b * 512 * 256;

    // ---- Phase 1: S = (Q K^T) * scale, masked, into LDS ----
    FragBF aq[2];
#pragma unroll
    for (int t = 0; t < 2; ++t) {
        const uint32_t* ap = Qp + (size_t)(lq0 + mlane) * 256 + ((hc + t * 32 + kbA) >> 1);
        aq[t].q[0] = *(const uint4*)ap;
        aq[t].q[1] = *(const uint4*)(ap + 4);
    }
    for (int n0 = 0; n0 < 512; n0 += 16) {
        v8f sacc = {};
#pragma unroll
        for (int t = 0; t < 2; ++t) {
            FragBF bk;
            const uint32_t* bp = Kp + (size_t)(n0 + mlane) * 256 + ((hc + t * 32 + kbB) >> 1);
            bk.q[0] = *(const uint4*)bp;
            bk.q[1] = *(const uint4*)(bp + 4);
            sacc = wmma_bf16(aq[t], bk, sacc);
        }
        int col = n0 + mlane;
        bool msk = mask[b * LL + col] != 0;
        int rowbase = wave * 16 + hi * 8;
#pragma unroll
        for (int r = 0; r < 8; ++r) {
            float v = msk ? -1e30f : sacc[r] * 0.125f;   // 1/sqrt(64)
            sS[(rowbase + r) * 512 + col] = v;
        }
    }
    __syncthreads();

    // ---- Phase 2: rowwise softmax, write bf16 probs in place ----
    if (threadIdx.x < 32) {
        float* row = sS + threadIdx.x * 512;
        float m = -3.4e38f;
        for (int j = 0; j < 512; ++j) m = fmaxf(m, row[j]);
        float s = 0.f;
        for (int j = 0; j < 512; ++j) s += __expf(row[j] - m);
        float inv = 1.0f / s;
        uint16_t* p = (uint16_t*)row;                // in-place compaction (write lags read)
        for (int j = 0; j < 512; ++j) {
            float e = __expf(row[j] - m) * inv;
            p[j] = f2bf(e);
        }
    }
    __syncthreads();

    // ---- Phase 3: O = P @ V  (A from LDS, B from v^T) ----
    v8f o[4] = {{}, {}, {}, {}};
    const uint32_t* Sp = (const uint32_t*)sS;        // row stride 512 u32 (2048 B)
    for (int kk = 0; kk < 512; kk += 32) {
        FragBF a;
        const uint32_t* ap = Sp + (size_t)(wave * 16 + mlane) * 512 + ((kk + kbA) >> 1);
        a.q[0] = *(const uint4*)ap;
        a.q[1] = *(const uint4*)(ap + 4);
#pragma unroll
        for (int t = 0; t < 4; ++t) {
            FragBF bv;
            const uint32_t* bp = Vp + (size_t)(hc + t * 16 + mlane) * 256 + ((kk + kbB) >> 1);
            bv.q[0] = *(const uint4*)bp;
            bv.q[1] = *(const uint4*)(bp + 4);
            o[t] = wmma_bf16(a, bv, o[t]);
        }
    }
#pragma unroll
    for (int t = 0; t < 4; ++t) {
        int cc = hc + t * 16 + mlane;
        size_t base = ((size_t)b * 512 + lq0 + hi * 8) * 512 + cc;
#pragma unroll
        for (int r = 0; r < 8; ++r) attnbf[base + (size_t)r * 512] = f2bf(o[t][r]);
    }
}

// ---------------------------------------------------------------------------
// Host orchestration
// ---------------------------------------------------------------------------
extern "C" void kernel_launch(void* const* d_in, const int* in_sizes, int n_in,
                              void* d_out, int out_size, void* d_ws, size_t ws_size,
                              hipStream_t stream) {
    (void)in_sizes; (void)n_in; (void)out_size; (void)ws_size;

    const float*   x      = (const float*)d_in[0];
    const uint8_t* mask   = (const uint8_t*)d_in[1];
    const float*   dw_w   = (const float*)d_in[2];
    const float*   dw_b   = (const float*)d_in[3];
    const float*   pw_w   = (const float*)d_in[4];
    const float*   pw_b   = (const float*)d_in[5];
    const float*   Wq     = (const float*)d_in[6];
    const float*   bq     = (const float*)d_in[7];
    const float*   Wk     = (const float*)d_in[8];
    const float*   bk     = (const float*)d_in[9];
    const float*   Wv     = (const float*)d_in[10];
    const float*   bv     = (const float*)d_in[11];
    const float*   Wo     = (const float*)d_in[12];
    const float*   bo     = (const float*)d_in[13];
    const float*   Wfc    = (const float*)d_in[14];
    const float*   bfc    = (const float*)d_in[15];
    const float*   normbw = (const float*)d_in[16];
    const float*   normbb = (const float*)d_in[17];
    const float*   normsw = (const float*)d_in[18];
    const float*   normsb = (const float*)d_in[19];
    const float*   normew = (const float*)d_in[20];
    const float*   normeb = (const float*)d_in[21];
    float* out = (float*)d_out;

    // Workspace layout
    const size_t SZF = (size_t)BCL * 4;   // 33 554 432 B
    char* w = (char*)d_ws;
    float*    r0      = (float*)w;
    float*    r1      = (float*)(w + SZF);
    float*    lnbuf   = (float*)(w + 2 * SZF);
    uint16_t* xbf     = (uint16_t*)(w + 3 * SZF);
    uint16_t* qbf     = xbf + BCL;
    uint16_t* kbf     = qbf + BCL;
    uint16_t* vbfT    = kbf + BCL;
    uint16_t* wbf     = vbfT + BCL;       // 9 x 512*512 bf16 weights
    uint16_t* attnbf  = xbf;              // alias: xbf free after QKV GEMMs

    uint16_t* wbf_pw = wbf;               // slots 0..3
    uint16_t* wbf_q  = wbf + 4 * CL;
    uint16_t* wbf_k  = wbf + 5 * CL;
    uint16_t* wbf_v  = wbf + 6 * CL;
    uint16_t* wbf_o  = wbf + 7 * CL;
    uint16_t* wbf_fc = wbf + 8 * CL;

    const dim3 gEW((BCL + 255) / 256), bEW(256);
    const dim3 gGemm(8, 8, BB), bGemm(256);
    const dim3 gAttn(16, HH, BB), bAttn(64);

    // Weight conversion to bf16 (row-major preserved)
    f32_to_bf16_kernel<<<dim3((4 * CL + 255) / 256), bEW, 0, stream>>>(pw_w, wbf_pw, 4 * CL);
    f32_to_bf16_kernel<<<dim3((CL + 255) / 256), bEW, 0, stream>>>(Wq,  wbf_q,  CL);
    f32_to_bf16_kernel<<<dim3((CL + 255) / 256), bEW, 0, stream>>>(Wk,  wbf_k,  CL);
    f32_to_bf16_kernel<<<dim3((CL + 255) / 256), bEW, 0, stream>>>(Wv,  wbf_v,  CL);
    f32_to_bf16_kernel<<<dim3((CL + 255) / 256), bEW, 0, stream>>>(Wo,  wbf_o,  CL);
    f32_to_bf16_kernel<<<dim3((CL + 255) / 256), bEW, 0, stream>>>(Wfc, wbf_fc, CL);

    // out = x + posenc ; res = out ; out = LN_b(out)
    posadd_kernel<<<gEW, bEW, 0, stream>>>(x, r0);
    ln2d_kernel<<<dim3(BB), bEW, 0, stream>>>(r0, normbw, normbb, lnbuf);

    // 4 depthwise-separable conv layers
    float* res = r0;
    float* nxt = r1;
    for (int i = 0; i < NCONV; ++i) {
        dwconv_kernel<<<gEW, bEW, 0, stream>>>(lnbuf, dw_w + (size_t)i * CC * KW,
                                               dw_b + (size_t)i * CC, xbf);
        gemm512_kernel<0><<<gGemm, bGemm, 0, stream>>>(wbf_pw + (size_t)i * CL, xbf,
                                                       pw_b + (size_t)i * CC, res, nxt, nullptr);
        ln2d_kernel<<<dim3(BB), bEW, 0, stream>>>(nxt, normsw + (size_t)i * CL,
                                                  normsb + (size_t)i * CL, lnbuf);
        float* t = res; res = nxt; nxt = t;
    }

    // Attention
    tconv_kernel<<<gEW, bEW, 0, stream>>>(lnbuf, xbf);
    gemm512_kernel<2><<<gGemm, bGemm, 0, stream>>>(wbf_q, xbf, bq, nullptr, nullptr, qbf);
    gemm512_kernel<2><<<gGemm, bGemm, 0, stream>>>(wbf_k, xbf, bk, nullptr, nullptr, kbf);
    gemm512_kernel<3><<<gGemm, bGemm, 0, stream>>>(wbf_v, xbf, bv, nullptr, nullptr, vbfT);
    attn_kernel<<<gAttn, bAttn, 0, stream>>>(qbf, kbf, vbfT, mask, attnbf);
    gemm512_kernel<1><<<gGemm, bGemm, 0, stream>>>(wbf_o, attnbf, bo, res, nxt, nullptr);
    ln2d_kernel<<<dim3(BB), bEW, 0, stream>>>(nxt, normew, normeb, lnbuf);
    res = nxt;

    // Final FC: out = relu(Wfc @ ln + bfc) + res
    tconv_kernel<<<gEW, bEW, 0, stream>>>(lnbuf, xbf);
    gemm512_kernel<0><<<gGemm, bGemm, 0, stream>>>(wbf_fc, xbf, bfc, res, out, nullptr);
}
